// MambaExpert_11424613007414
// MI455X (gfx1250) — matrix operand order repaired and checked by reading, not compiled, not run on hardware
//
#include <hip/hip_runtime.h>

// ---------------- problem constants (from reference) ----------------
#define L_SEQ 4096           // H*W
#define CCH   192            // C
#define DIN   384            // d_inner
#define DST   16             // d_state
#define DTR   12             // dt_rank
#define NPAD  48             // dt_rank + 2*d_state = 44, padded to 48 for WMMA
#define MROWS 8192           // B * L

typedef __attribute__((ext_vector_type(16))) __bf16 v16bf;
typedef __attribute__((ext_vector_type(8)))  __bf16 v8bf;
typedef __attribute__((ext_vector_type(8)))  float  v8f;

// ---------------- weight conversion ----------------
__global__ void k_cvt_bf16(const float* __restrict__ src, __bf16* __restrict__ dst, int n) {
    int i = blockIdx.x * 256 + threadIdx.x;
    if (i < n) dst[i] = (__bf16)src[i];
}

// x_proj_w is (44, 384); pad to (48, 384) with zero rows
__global__ void k_cvt_pad_xproj(const float* __restrict__ src, __bf16* __restrict__ dst) {
    int i = blockIdx.x * 256 + threadIdx.x;
    if (i >= NPAD * DIN) return;
    int r = i / DIN, k = i % DIN;
    dst[i] = (__bf16)((r < 44) ? src[r * DIN + k] : 0.0f);
}

// ---------------- LayerNorm + NCHW -> (M, C) transpose, emit bf16 ----------------
__global__ void k_ln(const float* __restrict__ x, const float* __restrict__ g,
                     const float* __restrict__ be, __bf16* __restrict__ xn) {
    int row = blockIdx.x * 64 + threadIdx.x;      // [0, MROWS)
    int b = row >> 12, l = row & (L_SEQ - 1);
    const float* xb = x + (size_t)b * CCH * L_SEQ + l;
    float s = 0.f, ss = 0.f;
    for (int c = 0; c < CCH; ++c) {
        float v = xb[(size_t)c * L_SEQ];
        s += v; ss += v * v;
    }
    float mu  = s * (1.0f / CCH);
    float var = ss * (1.0f / CCH) - mu * mu;
    float inv = rsqrtf(var + 1e-5f);
    __bf16* dst = xn + (size_t)row * CCH;
    for (int c = 0; c < CCH; ++c) {
        float v = xb[(size_t)c * L_SEQ];
        dst[c] = (__bf16)(((v - mu) * inv) * g[c] + be[c]);
    }
}

// ---------------- blocked WMMA GEMM: C[M,N] = A[M,K] * W[N,K]^T ----------------
// Block = 256 threads = 8 waves, covers 128 rows x (NB*16) cols.
// Weight tile (NB*16 rows x K, contiguous in N-major W) is async-staged into LDS
// once per block (global_load_async_to_lds_b128 / ASYNCcnt), then all 8 waves
// read B fragments from LDS. Per k-step: load A frag + all NB B frags, THEN
// issue NB v_wmma_f32_16x16x32_bf16 reusing the single A fragment, so one
// dscnt wait covers the batch and loads overlap the matrix ops.
// Per-lane layouts used here (contiguous K order, same for A and B):
//   A 16x32 bf16: lane -> row M=lane&15, 16 consecutive K from (lane>>4)*16
//   B 32x16 bf16: lane -> col N=lane&15, 16 consecutive K from (lane>>4)*16
//   D f32: VGPR r -> row r + (lane>>4)*8, col = lane&15
// EPI: 0 = store f32, 1 = store bf16, 2 = +bias then transpose to (B,C,H*W) f32.
template<int NB, int EPI>
__global__ void k_gemm_wmma(const __bf16* __restrict__ A, const __bf16* __restrict__ W,
                            int N, int K, int nColBlocks,
                            float* __restrict__ f32out, __bf16* __restrict__ bf16out,
                            const float* __restrict__ bias, float* __restrict__ finalOut) {
    extern __shared__ char smemRaw[];
    __bf16* Wt = (__bf16*)smemRaw;
    const int NBLK = NB * 16;
    int colBlock = blockIdx.x % nColBlocks;
    int rowBlock = blockIdx.x / nColBlocks;
    int tid = threadIdx.x;

    // ---- async stage of contiguous weight tile into LDS ----
    {
        const char* src = (const char*)(W + (size_t)colBlock * NBLK * K);
        int nBytes = NBLK * K * 2;                      // multiple of 4096 -> uniform loop
        for (int o = tid * 16; o < nBytes; o += 256 * 16) {
            unsigned ldsOff = (unsigned)(size_t)(smemRaw + o);
            asm volatile("global_load_async_to_lds_b128 %0, %1, off"
                         :: "v"(ldsOff), "v"(src + o) : "memory");
        }
        asm volatile("s_wait_asynccnt 0x0" ::: "memory");
        __syncthreads();
    }

    int wave = tid >> 5, lane = tid & 31;
    int tm = rowBlock * 8 + wave;
    int hi = lane >> 4;
    int ln15 = lane & 15;
    const __bf16* Arow = A + (size_t)(tm * 16 + ln15) * K;
    int ka = hi * 16, kb = hi * 16;

    v8f acc[NB];
#pragma unroll
    for (int nb = 0; nb < NB; ++nb) acc[nb] = (v8f){0.f,0.f,0.f,0.f,0.f,0.f,0.f,0.f};

    for (int k0 = 0; k0 < K; k0 += 32) {
        // A fragment: one aligned 32B load, 16 consecutive K
        v16bf a = *(const v16bf*)(Arow + k0 + ka);
        // all B fragments first (aligned 32B LDS loads) ...
        v16bf bfrag[NB];
#pragma unroll
        for (int nb = 0; nb < NB; ++nb) {
            bfrag[nb] = *(const v16bf*)(Wt + (size_t)(nb * 16 + ln15) * K + k0 + kb);
        }
        // ... then all WMMAs, reusing the single A fragment
#pragma unroll
        for (int nb = 0; nb < NB; ++nb) {
            acc[nb] = __builtin_amdgcn_wmma_f32_16x16x32_bf16(
                false, a, false, bfrag[nb], (short)0, acc[nb], false, false);
        }
    }

#pragma unroll
    for (int nb = 0; nb < NB; ++nb) {
        int col = colBlock * NBLK + nb * 16 + ln15;
#pragma unroll
        for (int r = 0; r < 8; ++r) {
            int rowM = tm * 16 + hi * 8 + r;
            float v = acc[nb][r];
            if (EPI == 0) {
                f32out[(size_t)rowM * N + col] = v;
            } else if (EPI == 1) {
                bf16out[(size_t)rowM * N + col] = (__bf16)v;
            } else {
                int b = rowM >> 12, l = rowM & (L_SEQ - 1);
                finalOut[((size_t)b * CCH + col) * L_SEQ + l] = v + bias[col];
            }
        }
    }
}

// ---------------- depthwise causal conv (k=4) + bias + SiLU ----------------
__global__ void k_conv(const float* __restrict__ xz, const float* __restrict__ cw,
                       const float* __restrict__ cb, float* __restrict__ xiC,
                       __bf16* __restrict__ xiCb) {
    int idx = blockIdx.x * 256 + threadIdx.x;
    if (idx >= MROWS * DIN) return;
    int row = idx / DIN, d = idx - row * DIN;
    int l = row & (L_SEQ - 1);
    float acc = cb[d];
#pragma unroll
    for (int j = 0; j < 4; ++j) {
        int ll = l - 3 + j;
        if (ll >= 0) acc += cw[d * 4 + j] * xz[(size_t)(row - 3 + j) * 768 + d];
    }
    float s = acc * (1.0f / (1.0f + __expf(-acc)));   // SiLU
    xiC[idx]  = s;
    xiCb[idx] = (__bf16)s;
}

// ---------------- delta = softplus(x_dbl[:,:12] @ dt_w^T + dt_b) ----------------
__global__ void k_dt(const float* __restrict__ xdbl, const float* __restrict__ dtw,
                     const float* __restrict__ dtb, float* __restrict__ delta) {
    int idx = blockIdx.x * 256 + threadIdx.x;
    if (idx >= MROWS * DIN) return;
    int row = idx / DIN, d = idx - row * DIN;
    const float* xr = xdbl + (size_t)row * NPAD;
    float acc = dtb[d];
#pragma unroll
    for (int r = 0; r < DTR; ++r) acc += xr[r] * dtw[d * DTR + r];
    delta[idx] = (acc > 20.f) ? acc : log1pf(__expf(acc));
}

// ---------------- selective scan, 8 lanes per channel (2 states each) ----------------
// t = ((b*DIN + d)*8 + g): lanes g=0..7 of one channel sit in one wave32;
// y = h·C reduced with width-8 shfl_xor; g==0 applies +xi*D, SiLU(z) gate, store.
__global__ void k_scan(const float* __restrict__ delta, const float* __restrict__ xdbl,
                       const float* __restrict__ xiC, const float* __restrict__ xz,
                       const float* __restrict__ Alog, const float* __restrict__ Dp,
                       __bf16* __restrict__ ybf) {
    int t = blockIdx.x * 256 + threadIdx.x;        // [0, B*DIN*8)
    int g = t & 7;
    int rem = t >> 3;                               // b*DIN + d
    int d = rem % DIN;
    int b = rem / DIN;
    int n0 = g * 2;

    float Ar0 = -__expf(Alog[d * DST + n0]);
    float Ar1 = -__expf(Alog[d * DST + n0 + 1]);
    float h0 = 0.f, h1 = 0.f;
    float Dd = Dp[d];

    for (int l = 0; l < L_SEQ; ++l) {
        size_t row = (size_t)b * L_SEQ + l;
        float dlt = delta[row * DIN + d];
        float xv  = xiC[row * DIN + d];
        const float* bc = xdbl + row * NPAD + DTR;   // [0..15]=B_t, [16..31]=C_t
        float sB0 = bc[n0],      sB1 = bc[n0 + 1];
        float sC0 = bc[16 + n0], sC1 = bc[17 + n0];
        h0 = __expf(dlt * Ar0) * h0 + dlt * sB0 * xv;
        h1 = __expf(dlt * Ar1) * h1 + dlt * sB1 * xv;
        float y = h0 * sC0 + h1 * sC1;
        y += __shfl_xor(y, 1, 8);
        y += __shfl_xor(y, 2, 8);
        y += __shfl_xor(y, 4, 8);
        if (g == 0) {
            float yt = y + xv * Dd;
            float z  = xz[row * 768 + DIN + d];
            float gt = z * (1.0f / (1.0f + __expf(-z)));
            ybf[row * DIN + d] = (__bf16)(yt * gt);
        }
    }
}

// ---------------- launcher ----------------
extern "C" void kernel_launch(void* const* d_in, const int* in_sizes, int n_in,
                              void* d_out, int out_size, void* d_ws, size_t ws_size,
                              hipStream_t stream) {
    const float* x       = (const float*)d_in[0];
    const float* ln_g    = (const float*)d_in[1];
    const float* ln_b    = (const float*)d_in[2];
    const float* in_w    = (const float*)d_in[3];   // (768, 192)
    const float* conv_w  = (const float*)d_in[4];   // (384, 1, 4)
    const float* conv_b  = (const float*)d_in[5];
    const float* xproj_w = (const float*)d_in[6];   // (44, 384)
    const float* dt_w    = (const float*)d_in[7];   // (384, 12)
    const float* dt_b    = (const float*)d_in[8];
    const float* A_log   = (const float*)d_in[9];   // (384, 16)
    const float* Dp      = (const float*)d_in[10];
    const float* out_w   = (const float*)d_in[11];  // (192, 384)
    const float* proj_w  = (const float*)d_in[12];  // (192, 192)
    const float* proj_b  = (const float*)d_in[13];
    float* out = (float*)d_out;

    // workspace layout (256B aligned)
    char* ws = (char*)d_ws;
    size_t off = 0;
    auto alloc = [&](size_t bytes) { char* p = ws + off; off = (off + bytes + 255) & ~(size_t)255; return p; };
    __bf16* xn_bf   = (__bf16*)alloc((size_t)MROWS * CCH * 2);
    __bf16* w_in_bf = (__bf16*)alloc((size_t)768 * CCH * 2);
    __bf16* w_xp_bf = (__bf16*)alloc((size_t)NPAD * DIN * 2);
    __bf16* w_ow_bf = (__bf16*)alloc((size_t)CCH * DIN * 2);
    __bf16* w_pj_bf = (__bf16*)alloc((size_t)CCH * CCH * 2);
    float*  xz_f    = (float*) alloc((size_t)MROWS * 768 * 4);
    float*  xiC_f   = (float*) alloc((size_t)MROWS * DIN * 4);
    __bf16* xiC_bf  = (__bf16*)alloc((size_t)MROWS * DIN * 2);
    float*  xdbl_f  = (float*) alloc((size_t)MROWS * NPAD * 4);
    float*  delta_f = (float*) alloc((size_t)MROWS * DIN * 4);
    __bf16* y_bf    = (__bf16*)alloc((size_t)MROWS * DIN * 2);
    __bf16* y2_bf   = (__bf16*)alloc((size_t)MROWS * CCH * 2);
    (void)ws_size; (void)in_sizes; (void)n_in; (void)out_size;

    // 1) weight conversions
    k_cvt_bf16<<<(768 * CCH + 255) / 256, 256, 0, stream>>>(in_w, w_in_bf, 768 * CCH);
    k_cvt_pad_xproj<<<(NPAD * DIN + 255) / 256, 256, 0, stream>>>(xproj_w, w_xp_bf);
    k_cvt_bf16<<<(CCH * DIN + 255) / 256, 256, 0, stream>>>(out_w, w_ow_bf, CCH * DIN);
    k_cvt_bf16<<<(CCH * CCH + 255) / 256, 256, 0, stream>>>(proj_w, w_pj_bf, CCH * CCH);

    // 2) layernorm + transpose
    k_ln<<<MROWS / 64, 64, 0, stream>>>(x, ln_g, ln_b, xn_bf);

    // 3) in_proj GEMM: (8192 x 192) * (768 x 192)^T -> xz f32
    {
        int nCB = 768 / 64;   // 12
        size_t sh = (size_t)64 * CCH * 2;
        k_gemm_wmma<4, 0><<<(MROWS / 128) * nCB, 256, sh, stream>>>(
            xn_bf, w_in_bf, 768, CCH, nCB, xz_f, nullptr, nullptr, nullptr);
    }

    // 4) depthwise conv + SiLU
    k_conv<<<(MROWS * DIN + 255) / 256, 256, 0, stream>>>(xz_f, conv_w, conv_b, xiC_f, xiC_bf);

    // 5) x_proj GEMM: (8192 x 384) * (48 x 384)^T -> x_dbl f32 (padded)
    {
        int nCB = 1;          // 48 cols = one NB=3 block
        size_t sh = (size_t)48 * DIN * 2;
        k_gemm_wmma<3, 0><<<(MROWS / 128) * nCB, 256, sh, stream>>>(
            xiC_bf, w_xp_bf, NPAD, DIN, nCB, xdbl_f, nullptr, nullptr, nullptr);
    }

    // 6) delta = softplus(low-rank GEMV + bias)
    k_dt<<<(MROWS * DIN + 255) / 256, 256, 0, stream>>>(xdbl_f, dt_w, dt_b, delta_f);

    // 7) sequential selective scan + gate -> y bf16  (8 lanes per channel)
    k_scan<<<(2 * DIN * 8) / 256, 256, 0, stream>>>(delta_f, xdbl_f, xiC_f, xz_f, A_log, Dp, y_bf);

    // 8) out_proj GEMM: (8192 x 384) * (192 x 384)^T -> y2 bf16
    {
        int nCB = 192 / 64;   // 3
        size_t sh = (size_t)64 * DIN * 2;
        k_gemm_wmma<4, 1><<<(MROWS / 128) * nCB, 256, sh, stream>>>(
            y_bf, w_ow_bf, CCH, DIN, nCB, nullptr, y2_bf, nullptr, nullptr);
    }

    // 9) final proj GEMM + bias + transpose to (B, C, H, W)
    {
        int nCB = 192 / 64;   // 3
        size_t sh = (size_t)64 * CCH * 2;
        k_gemm_wmma<4, 2><<<(MROWS / 128) * nCB, 256, sh, stream>>>(
            y2_bf, w_pj_bf, CCH, CCH, nCB, nullptr, nullptr, proj_b, out);
    }
}
